// EqxPhACE_90993177133601
// MI455X (gfx1250) — matrix operand order repaired and checked by reading, not compiled, not run on hardware
//
#include <hip/hip_runtime.h>
#include <hip/hip_bf16.h>
#include <math.h>

// ---------------- problem constants (from reference) ----------------
#define NPAIRS   200000
#define NATOMS   5000
#define NPROBE   18          // int((5.0-0.5)/0.25)
#define NBASIS   32
#define NKNOTS   64
#define EMBD     64
#define KTOT     160         // 64+48+32+16
#define DENSW    480         // sum (2l+1)*K_l
#define SPLMAX   5.5f        // MAX_CUTOFF*1.1
#define INV2PI   0.15915494309189535f

__device__ __constant__ int c_KL[4]  = {64, 48, 32, 16};
__device__ __constant__ int c_KO[4]  = {0, 64, 112, 144};   // k offsets in packed 160
__device__ __constant__ int c_OFF[4] = {0, 64, 208, 368};   // (l,m,k) offsets in 480

struct Ptrs4 { const float* p[4]; };

typedef __attribute__((ext_vector_type(16))) _Float16 v16h;
typedef __attribute__((ext_vector_type(8)))  float    v8f;
typedef __attribute__((ext_vector_type(4)))  unsigned int v4u;
typedef __attribute__((ext_vector_type(8)))  int          v8i;
typedef __attribute__((ext_vector_type(4)))  int          v4i;

#ifndef __has_builtin
#define __has_builtin(x) 0
#endif
#if __has_builtin(__builtin_amdgcn_tensor_load_to_lds)
#define USE_TDM 1
#else
#define USE_TDM 0
#endif

__device__ inline void wave_sync() {
  __builtin_amdgcn_wave_barrier();
  asm volatile("" ::: "memory");
}

// single v_rcp_f32 instead of the IEEE divide expansion
__device__ inline float fast_rcp(float x) { return __builtin_amdgcn_rcpf(x); }

// fast tanh via HW exp: inf-safe (e^inf -> 1, e^-inf -> -1)
__device__ inline float fast_tanh(float u) {
  return 1.f - 2.f * fast_rcp(__expf(2.f * u) + 1.f);
}
__device__ inline float fast_silu(float x) {
  return x * fast_rcp(1.f + __expf(-x));
}

// smooth part of the cutoff, only valid for 0 < scaled < 1
__device__ inline float cutoff_mid(float scaled) {
  float ss = fminf(fmaxf(scaled, 1e-10f), 1.f - 1e-10f);
  float u = fast_rcp(__tanf(3.14159265358979f * ss));
  return 0.5f * (1.f + fast_tanh(u));
}
__device__ inline float cutoff_f(float r, float c, float width) {
  float scaled = (r - (c - width)) / width;
  return scaled <= 0.f ? 1.f : (scaled >= 1.f ? 0.f : cutoff_mid(scaled));
}

// ---- WMMA operand staging per ISA 7.12.2 (wave32) ----
// A (16x32 f16, row-major in LDS, row stride ldk):
//   lanes 0-15: M=lane, K = {0..7, 16..23}; lanes 16-31: M=lane-16, K={8..15,24..31}
__device__ inline v16h load_A_f16(const _Float16* lds, int ldk, int lane) {
  int m = lane & 15;
  int kh = (lane >> 4) & 1;
  const _Float16* row = lds + m * ldk;
  v16h a;
#pragma unroll
  for (int v = 0; v < 8; ++v) {
    int kb = (v < 4) ? (2 * v) : (16 + 2 * (v - 4));
    int k = kb + kh * 8;
    a[2 * v]     = row[k];
    a[2 * v + 1] = row[k + 1];
  }
  return a;
}
// B (32x16 f16, row-major KxN in LDS, row stride ldn):
//   lanes 0-15: N=lane, K = {0..7, 16..23}; lanes 16-31: N=lane-16, K={8..15,24..31}
__device__ inline v16h load_B_f16(const _Float16* lds, int ldn, int lane) {
  int n = lane & 15;
  int kh = (lane >> 4) & 1;
  v16h b;
#pragma unroll
  for (int v = 0; v < 8; ++v) {
    int k = ((v < 4) ? (2 * v) : (16 + 2 * (v - 4))) + kh * 8;
    b[2 * v]     = lds[k * ldn + n];
    b[2 * v + 1] = lds[(k + 1) * ldn + n];
  }
  return b;
}

#if USE_TDM
// Tensor Data Mover: 2D f32 tile (dim0 fastest, contiguous rows of `stride0`)
// from global -> LDS. D# layout per CDNA5 ISA.
// This toolchain exposes the 6-arg builtin:
//   (uint32x4 g0, int32x8 g1, int32x4 g2, int32x4 g3, int32x8 g4, i32 cpol)
__device__ inline void tdm_load_2d_f32(unsigned lds_off, const void* gptr,
                                       unsigned dim0, unsigned dim1,
                                       unsigned long long stride0) {
  union { unsigned long long q[2]; v4u v; } g0;
  union { unsigned long long q[4]; v8i v; } g1, g4;
  union { unsigned long long q[2]; v4i v; } g2, g3;
  unsigned long long ga = (unsigned long long)(uintptr_t)gptr;
  // group0: count=1 | lds_addr[63:32] ; global_addr[120:64], type=2 [127:126]
  g0.q[0] = 1ull | ((unsigned long long)lds_off << 32);
  g0.q[1] = (ga & ((1ull << 57) - 1ull)) | (2ull << 62);
  // group1: data_size=2 (4B) [17:16]; tensor_dim0 [79:48]; tensor_dim1 [111:80];
  //         tile_dim0 [127:112]; tile_dim1 [143:128]; dim0_stride [207:160]
  g1.q[0] = (2ull << 16) | (((unsigned long long)dim0 & 0xFFFFull) << 48);
  g1.q[1] = (((unsigned long long)dim0 >> 16) & 0xFFFFull)
          | (((unsigned long long)dim1 & 0xFFFFFFFFull) << 16)
          | (((unsigned long long)dim0 & 0xFFFFull) << 48);
  g1.q[2] = ((unsigned long long)dim1 & 0xFFFFull)
          | ((stride0 & 0xFFFFFFFFull) << 32);
  g1.q[3] = (stride0 >> 32) & 0xFFFFull;
  g2.q[0] = 0; g2.q[1] = 0;
  g3.q[0] = 0; g3.q[1] = 0;
  g4.q[0] = 0; g4.q[1] = 0; g4.q[2] = 0; g4.q[3] = 0;
  __builtin_amdgcn_tensor_load_to_lds(g0.v, g1.v, g2.v, g3.v, g4.v, 0);
}
#endif

// =====================================================================
// Kernel A: pair geometry, spherical harmonics (l<=3), probe histogram
// =====================================================================
__global__ void k_pairs_geom(const float* __restrict__ pos,
                             const float* __restrict__ cells,
                             const int* __restrict__ shifts,
                             const int* __restrict__ ctr,
                             const int* __restrict__ nbr,
                             const int* __restrict__ spair,
                             float* __restrict__ r_out,
                             float* __restrict__ sh_out,
                             float* __restrict__ pnn) {
  int p = blockIdx.x * blockDim.x + threadIdx.x;
  if (p >= NPAIRS) return;
  int c = ctr[p], nb = nbr[p], sp = spair[p];
  float sxf = (float)shifts[3 * p + 0];
  float syf = (float)shifts[3 * p + 1];
  float szf = (float)shifts[3 * p + 2];
  const float* C = cells + 9 * sp;
  float vx = pos[3 * nb + 0] - pos[3 * c + 0] + sxf * C[0] + syf * C[3] + szf * C[6];
  float vy = pos[3 * nb + 1] - pos[3 * c + 1] + sxf * C[1] + syf * C[4] + szf * C[7];
  float vz = pos[3 * nb + 2] - pos[3 * c + 2] + sxf * C[2] + syf * C[5] + szf * C[8];
  float rsq = vx * vx + vy * vy + vz * vz;
  float r = sqrtf(rsq);
  r_out[p] = r;

  float inv = __builtin_amdgcn_rsqf(rsq > 0.f ? rsq : 1.f);
  float x = vx * inv, y = vy * inv, z = vz * inv;

  float Q1 = z, Q2 = -1.f;
  float Q3 = 0.5f * (3.f * z * z - 1.f);
  float Q4 = -3.f * z, Q5 = 3.f;
  float Q6 = (5.f * z * Q3 - 2.f * z) * (1.f / 3.f);
  float Q7 = 1.5f - 7.5f * z * z;
  float Q8 = 15.f * z, Q9 = -15.f;
  float s1 = y, c1 = x;
  float s2 = x * s1 + y * c1, c2 = x * c1 - y * s1;
  float s3 = x * s2 + y * c2, c3 = x * c2 - y * s2;
  const float F0 = sqrtf(1.f * INV2PI);
  const float F1 = sqrtf(3.f * INV2PI);
  const float F2 = sqrtf(1.5f * INV2PI);
  const float F3 = sqrtf(5.f * INV2PI);
  const float F4 = sqrtf(5.f / 6.f * INV2PI);
  const float F5 = sqrtf(5.f / 24.f * INV2PI);
  const float F6 = sqrtf(7.f * INV2PI);
  const float F7 = sqrtf(7.f / 12.f * INV2PI);
  const float F8 = sqrtf(7.f / 120.f * INV2PI);
  const float F9 = sqrtf(7.f / 720.f * INV2PI);
  const float isq2 = 0.7071067811865475f;
  float* Y = sh_out + 16 * p;
  Y[0]  = F0 * isq2;
  Y[1]  = F2 * Q2 * s1;  Y[2]  = F1 * Q1 * isq2;  Y[3]  = F2 * Q2 * c1;
  Y[4]  = F5 * Q5 * s2;  Y[5]  = F4 * Q4 * s1;    Y[6]  = F3 * Q3 * isq2;
  Y[7]  = F4 * Q4 * c1;  Y[8]  = F5 * Q5 * c2;
  Y[9]  = F9 * Q9 * s3;  Y[10] = F8 * Q8 * s2;    Y[11] = F7 * Q7 * s1;
  Y[12] = F6 * Q6 * isq2; Y[13] = F7 * Q7 * c1;   Y[14] = F8 * Q8 * c2;
  Y[15] = F9 * Q9 * c3;

  // Probe histogram. With spacing 0.25 and width 1.0 only ~4 probes land in
  // the smooth band; skip transcendentals (and the atomic for w==0) otherwise.
#pragma unroll
  for (int i = 0; i < NPROBE; ++i) {
    float pc = 0.5f + 0.25f * (float)i;
    float scaled = r + 1.f - pc;   // (r - (pc - 1)) / 1
    if (scaled >= 1.f) continue;   // w = 0
    float w = (scaled <= 0.f) ? 1.f : cutoff_mid(scaled);
    atomicAdd(&pnn[c * NPROBE + i], w);
  }
}

// =====================================================================
// Kernel B: per-atom adaptive cutoff (18 probes, gaussian soft-select)
// =====================================================================
__global__ void k_adaptive(const float* __restrict__ pnn, float* __restrict__ acut) {
  int a = blockIdx.x * blockDim.x + threadIdx.x;
  if (a >= NATOMS) return;
  float d[NPROBE], g[NPROBE];
#pragma unroll
  for (int i = 0; i < NPROBE; ++i) {
    float xb = (float)i / (float)(NPROBE - 1);
    d[i] = pnn[a * NPROBE + i] - 30.f + 30.f * xb * xb * xb;
  }
  g[0] = d[1] - d[0];
  g[NPROBE - 1] = d[NPROBE - 1] - d[NPROBE - 2];
#pragma unroll
  for (int i = 1; i < NPROBE - 1; ++i) g[i] = 0.5f * (d[i + 1] - d[i - 1]);
  float lw[NPROBE], mx = -1e30f;
#pragma unroll
  for (int i = 0; i < NPROBE; ++i) {
    float wt = fmaxf(fabsf(g[i]), 1e-12f);
    float q = d[i] * fast_rcp(wt);
    lw[i] = -0.5f * q * q;
    mx = fmaxf(mx, lw[i]);
  }
  float s = 0.f, acc = 0.f;
#pragma unroll
  for (int i = 0; i < NPROBE; ++i) { lw[i] = __expf(lw[i] - mx); s += lw[i]; }
#pragma unroll
  for (int i = 0; i < NPROBE; ++i) acc += lw[i] * (0.5f + 0.25f * (float)i);
  acut[a] = acc * fast_rcp(s);
}

// =====================================================================
// Kernel C: embN = RMSNorm(emb0)*w ; embW = embN @ W_in[l] (packed 160)
// =====================================================================
__global__ void k_embW(const float* __restrict__ emb0,
                       const float* __restrict__ rmsw,
                       Ptrs4 Win, float* __restrict__ embW) {
  int a = blockIdx.x;
  int t = threadIdx.x;
  __shared__ float se[EMBD];
  __shared__ float sn[EMBD];
  float e = emb0[a * EMBD + t];
  se[t] = e * e;
  __syncthreads();
  float s = 0.f;
#pragma unroll
  for (int j = 0; j < EMBD; ++j) s += se[j];
  float rinv = rsqrtf(s / (float)EMBD + 1e-5f);
  sn[t] = e * rinv * rmsw[t];
  __syncthreads();
  for (int k = t; k < KTOT; k += EMBD) {
    int l = (k < 64) ? 0 : (k < 112) ? 1 : (k < 144) ? 2 : 3;
    int kk = k - c_KO[l];
    int K = c_KL[l];
    const float* W = Win.p[l];
    float acc = 0.f;
#pragma unroll
    for (int j = 0; j < EMBD; ++j) acc += sn[j] * W[j * K + kk];
    embW[a * KTOT + k] = acc;
  }
}

// =====================================================================
// Kernel D: persistent-wave fused pair kernel (dynamic LDS, 64 KB).
//   TDM loads spline tables to LDS; rb spline*cw -> WMMA+silu -> WMMA
//   -> message scatter via global f32 atomics into dens.
// LDS layout (bytes):
//   [0      ) sW1h  4*32*64 f16 = 16384
//   [16384  ) sW2h  64*160 f16  = 20480
//   [36864  ) sSV   64*32 f32   =  8192
//   [45056  ) sSD   64*32 f32   =  8192
//   [53248  ) sRB   4 waves * 16*32 f16 = 4096
//   [57344  ) sHID  4 waves * 16*64 f16 = 8192   -> total 65536
// =====================================================================
#define LDS_W1   0
#define LDS_W2   16384
#define LDS_SV   36864
#define LDS_SD   45056
#define LDS_RB   53248
#define LDS_HID  57344
#define LDS_TOTAL 65536

__global__ __launch_bounds__(128) void k_pair_wmma(
    const float* __restrict__ r_in, const float* __restrict__ sh,
    const float* __restrict__ acut,
    const float* __restrict__ sv, const float* __restrict__ sd,
    Ptrs4 W1, Ptrs4 W2,
    const int* __restrict__ ctr, const int* __restrict__ nbr,
    const float* __restrict__ embW, float* __restrict__ dens) {
  extern __shared__ char smem[];
  _Float16* sW1 = (_Float16*)(smem + LDS_W1);
  _Float16* sW2 = (_Float16*)(smem + LDS_W2);
  float*    sSV = (float*)(smem + LDS_SV);
  float*    sSD = (float*)(smem + LDS_SD);

  const int tid = threadIdx.x;
  const int wid = tid >> 5;
  const int lane = tid & 31;
  _Float16* sRB  = (_Float16*)(smem + LDS_RB)  + wid * (16 * 32);
  _Float16* sHID = (_Float16*)(smem + LDS_HID) + wid * (16 * 64);

#if USE_TDM
  if (wid == 0) {
    // spline tables: 64 rows x 32 f32, contiguous (stride 32)
    tdm_load_2d_f32((unsigned)(uintptr_t)sSV, sv, NBASIS, NKNOTS, NBASIS);
    tdm_load_2d_f32((unsigned)(uintptr_t)sSD, sd, NBASIS, NKNOTS, NBASIS);
    __builtin_amdgcn_s_wait_tensorcnt(0);
  }
#else
  for (int i = tid; i < NKNOTS * NBASIS; i += 128) {
    sSV[i] = sv[i];
    sSD[i] = sd[i];
  }
#endif
  for (int i = tid; i < 4 * 2048; i += 128) {
    int l = i >> 11;
    sW1[i] = (_Float16)W1.p[l][i & 2047];
  }
  for (int i = tid; i < 64 * KTOT; i += 128) {
    int j = i / KTOT, cc = i - j * KTOT;
    int l = (cc < 64) ? 0 : (cc < 112) ? 1 : (cc < 144) ? 2 : 3;
    sW2[i] = (_Float16)W2.p[l][j * c_KL[l] + (cc - c_KO[l])];
  }
  __syncthreads();

  const int numTiles = (NPAIRS + 15) / 16;
  const float dx = SPLMAX / (float)(NKNOTS - 1);
  const float inv_dx = (float)(NKNOTS - 1) / SPLMAX;
  const v8f vzero = {0.f, 0.f, 0.f, 0.f, 0.f, 0.f, 0.f, 0.f};

  for (int tile = blockIdx.x * 4 + wid; tile < numTiles; tile += gridDim.x * 4) {
    const int p0 = tile * 16;
    // ---- stage rb (16 pairs x 32 basis, f16) ----
    {
      int pr = lane & 15;
      int half = lane >> 4;
      int p = p0 + pr;
      float rbv[16];
      if (p < NPAIRS) {
        float r = r_in[p];
        float cw = cutoff_f(r, acut[ctr[p]], 1.0f);
        int n = (int)floorf(r * inv_dx);
        n = n < 0 ? 0 : (n > NKNOTS - 2 ? NKNOTS - 2 : n);
        float t = (r - (float)n * dx) * inv_dx;
        float t2 = t * t, t3 = t2 * t;
        float h00 = 2.f * t3 - 3.f * t2 + 1.f;
        float h10 = (t3 - 2.f * t2 + t) * dx;
        float h01 = -2.f * t3 + 3.f * t2;
        float h11 = (t3 - t2) * dx;
        const float* v0 = sSV + n * NBASIS;
        const float* v1 = v0 + NBASIS;
        const float* d0 = sSD + n * NBASIS;
        const float* d1 = d0 + NBASIS;
#pragma unroll
        for (int j = 0; j < 16; ++j) {
          int cc = half * 16 + j;
          rbv[j] = cw * (h00 * v0[cc] + h10 * d0[cc] + h01 * v1[cc] + h11 * d1[cc]);
        }
      } else {
#pragma unroll
        for (int j = 0; j < 16; ++j) rbv[j] = 0.f;
      }
#pragma unroll
      for (int j = 0; j < 16; ++j)
        sRB[pr * 32 + half * 16 + j] = (_Float16)rbv[j];
    }
    wave_sync();
    const v16h a_rb = load_A_f16(sRB, 32, lane);

    const int n_col = lane & 15;
    const int mo = (lane >> 4) ? 8 : 0;

    for (int l = 0; l < 4; ++l) {
      // hidden = silu(rb @ W1[l]) : four 16-col tiles
      for (int nt = 0; nt < 4; ++nt) {
        v16h b = load_B_f16(sW1 + l * 2048 + nt * 16, 64, lane);
        v8f h = __builtin_amdgcn_wmma_f32_16x16x32_f16(
            false, a_rb, false, b, (short)0, vzero, false, false);
#pragma unroll
        for (int i = 0; i < 8; ++i)
          sHID[(mo + i) * 64 + nt * 16 + n_col] = (_Float16)fast_silu(h[i]);
      }
      wave_sync();
      // radial = hidden @ W2[l], scatter messages from accumulator layout
      const int nts = c_KL[l] / 16;
      const int K = c_KL[l];
      for (int nt = 0; nt < nts; ++nt) {
        v8f acc = vzero;
#pragma unroll
        for (int kk = 0; kk < 2; ++kk) {
          v16h ah = load_A_f16(sHID + kk * 32, 64, lane);
          v16h bh = load_B_f16(sW2 + (kk * 32) * KTOT + c_KO[l] + nt * 16, KTOT, lane);
          acc = __builtin_amdgcn_wmma_f32_16x16x32_f16(
              false, ah, false, bh, (short)0, acc, false, false);
        }
        int kloc = nt * 16 + n_col;
        int kglob = c_KO[l] + kloc;
#pragma unroll
        for (int i = 0; i < 8; ++i) {
          int p = p0 + mo + i;
          if (p < NPAIRS) {
            int c = ctr[p];
            int nb = nbr[p];
            float rv = acc[i] * embW[nb * KTOT + kglob];
            float* drow = dens + c * DENSW + c_OFF[l] + kloc;
            const float* shp = sh + p * 16 + l * l;
            for (int m = 0; m < 2 * l + 1; ++m)
              atomicAdd(drow + m * K, shp[m] * rv);
          }
        }
      }
      wave_sync();
    }
    wave_sync();
  }
}

// =====================================================================
// Kernel E: feats[l] = (dens * message_scaling) @ W_out[l]
// =====================================================================
__global__ void k_wout(const float* __restrict__ dens, Ptrs4 Wout,
                       const float* __restrict__ ms_p, float* __restrict__ feats) {
  int a = blockIdx.x;
  float ms = ms_p[0];
  for (int idx = threadIdx.x; idx < DENSW; idx += blockDim.x) {
    int l, m, j;
    if (idx < 64)        { l = 0; m = 0; j = idx; }
    else if (idx < 208)  { int r = idx - 64;  l = 1; m = r / 48; j = r - m * 48; }
    else if (idx < 368)  { int r = idx - 208; l = 2; m = r / 32; j = r - m * 32; }
    else                 { int r = idx - 368; l = 3; m = r / 16; j = r - m * 16; }
    int K = c_KL[l];
    const float* drow = dens + a * DENSW + c_OFF[l] + m * K;
    const float* W = Wout.p[l];
    float acc = 0.f;
    for (int k = 0; k < K; ++k) acc += drow[k] * W[k * K + j];
    feats[a * DENSW + c_OFF[l] + m * K + j] = ms * acc;
  }
}

// =====================================================================
// Kernel F: per group g: S = stack(feats k-slices), Unc = U[g]@S,
//   P = Unc^2, row0 of U[g]^T @ P feeds feats[0][0] + readout.
// =====================================================================
template <int G>
__device__ float group_term(const float* fa, int f, const float* Ug,
                            const float* Wro) {
  constexpr int NN = (G + 1) * (G + 1);
  float S[NN], Pv[NN];
#pragma unroll
  for (int lp = 0; lp <= G; ++lp) {
#pragma unroll
    for (int m = 0; m < 2 * lp + 1; ++m)
      S[lp * lp + m] = fa[c_OFF[lp] + m * c_KL[lp] + (3 - G) * 16 + f];
  }
#pragma unroll
  for (int row = 0; row < NN; ++row) {
    float acc = 0.f;
#pragma unroll
    for (int rr = 0; rr < NN; ++rr) acc += Ug[row * NN + rr] * S[rr];
    Pv[row] = acc * acc;
  }
  float c0 = 0.f;
#pragma unroll
  for (int rr = 0; rr < NN; ++rr) c0 += Ug[rr * NN] * Pv[rr];
  int k = G * 16 + f;
  return (fa[k] + c0) * Wro[k];
}

__global__ void k_final(const float* __restrict__ feats, Ptrs4 U,
                        const float* __restrict__ Wro, float* __restrict__ out) {
  int f = threadIdx.x & 15;
  int a = blockIdx.x * 8 + (threadIdx.x >> 4);
  if (a >= NATOMS) return;
  const float* fa = feats + a * DENSW;
  float res = group_term<0>(fa, f, U.p[0], Wro)
            + group_term<1>(fa, f, U.p[1], Wro)
            + group_term<2>(fa, f, U.p[2], Wro)
            + group_term<3>(fa, f, U.p[3], Wro);
#pragma unroll
  for (int o = 8; o > 0; o >>= 1) res += __shfl_xor(res, o, 16);
  if (f == 0) out[a] = res;
}

// =====================================================================
extern "C" void kernel_launch(void* const* d_in, const int* in_sizes, int n_in,
                              void* d_out, int out_size, void* d_ws, size_t ws_size,
                              hipStream_t stream) {
  const float* positions = (const float*)d_in[0];
  const float* cells     = (const float*)d_in[1];
  const float* emb0      = (const float*)d_in[2];
  const float* sv        = (const float*)d_in[3];
  const float* sd        = (const float*)d_in[4];
  const float* rmsw      = (const float*)d_in[5];
  const float* ms        = (const float*)d_in[6];
  Ptrs4 W1  {{(const float*)d_in[7],  (const float*)d_in[8],  (const float*)d_in[9],  (const float*)d_in[10]}};
  Ptrs4 W2  {{(const float*)d_in[11], (const float*)d_in[12], (const float*)d_in[13], (const float*)d_in[14]}};
  Ptrs4 Win {{(const float*)d_in[15], (const float*)d_in[16], (const float*)d_in[17], (const float*)d_in[18]}};
  Ptrs4 Wout{{(const float*)d_in[19], (const float*)d_in[20], (const float*)d_in[21], (const float*)d_in[22]}};
  Ptrs4 U   {{(const float*)d_in[23], (const float*)d_in[24], (const float*)d_in[25], (const float*)d_in[26]}};
  const float* Wro    = (const float*)d_in[27];
  const int* shifts   = (const int*)d_in[28];
  const int* ctr      = (const int*)d_in[29];
  const int* nbr      = (const int*)d_in[30];
  const int* spair    = (const int*)d_in[31];

  float* ws   = (float*)d_ws;
  float* r    = ws;                       // NPAIRS
  float* sh   = r + NPAIRS;               // 16*NPAIRS
  float* pnn  = sh + 16 * NPAIRS;         // NATOMS*NPROBE
  float* acut = pnn + NATOMS * NPROBE;    // NATOMS
  float* embW = acut + NATOMS;            // NATOMS*KTOT
  float* dens = embW + NATOMS * KTOT;     // NATOMS*DENSW
  float* feats = dens + NATOMS * DENSW;   // NATOMS*DENSW

  (void)hipMemsetAsync(pnn, 0, NATOMS * NPROBE * sizeof(float), stream);
  (void)hipMemsetAsync(dens, 0, NATOMS * DENSW * sizeof(float), stream);

  k_pairs_geom<<<(NPAIRS + 255) / 256, 256, 0, stream>>>(
      positions, cells, shifts, ctr, nbr, spair, r, sh, pnn);
  k_adaptive<<<(NATOMS + 127) / 128, 128, 0, stream>>>(pnn, acut);
  k_embW<<<NATOMS, EMBD, 0, stream>>>(emb0, rmsw, Win, embW);
  k_pair_wmma<<<256, 128, LDS_TOTAL, stream>>>(r, sh, acut, sv, sd, W1, W2,
                                               ctr, nbr, embW, dens);
  k_wout<<<NATOMS, 128, 0, stream>>>(dens, Wout, ms, feats);
  k_final<<<(NATOMS + 7) / 8, 128, 0, stream>>>(feats, U, Wro, (float*)d_out);
}
